// CausalSelfAttention_89094801588628
// MI455X (gfx1250) — matrix operand order repaired and checked by reading, not compiled, not run on hardware
//
#include <hip/hip_runtime.h>

// ---------------------------------------------------------------------------
// Causal self-attention for MI455X (gfx1250), bf16 WMMA pipeline.
//   B=4, T=2048, D=1024, H=16, Dh=64.  wave32, V_WMMA_F32_16X16X32_BF16.
//   Projection GEMMs stage the shared A tile into LDS with async-to-LDS
//   copies (ASYNCcnt path); B weights stream from L2 via global_load_b128.
// ---------------------------------------------------------------------------

#define D_MODEL   1024
#define N_HEADS   16
#define HEAD_DIM  64
#define BATCH     4
#define SEQ       2048
#define NTOK      (BATCH * SEQ)   // 8192 tokens

typedef __attribute__((ext_vector_type(16))) __bf16          bf16x16;
typedef __attribute__((ext_vector_type(8)))  __bf16          bf16x8;
typedef __attribute__((ext_vector_type(8)))  float           f32x8;
typedef __attribute__((ext_vector_type(8)))  unsigned short  u16x8;
typedef __attribute__((ext_vector_type(4)))  int             v4i;

__device__ __forceinline__ unsigned short f32_to_bf16(float f) {
  unsigned int u = __float_as_uint(f);
  u += 0x7FFFu + ((u >> 16) & 1u);          // round-to-nearest-even
  return (unsigned short)(u >> 16);
}

__device__ __forceinline__ f32x8 wmma_bf16(bf16x16 a, bf16x16 b, f32x8 c) {
  // D = A(16x32) * B(32x16) + C, fp32 accumulate
  return __builtin_amdgcn_wmma_f32_16x16x32_bf16(
      false, a, false, b, (short)0, c, false, false);
}

// A-fragment (16x32 bf16, MxK): lane<16 holds row=lane, K = {0..7, 16..23};
// lane>=16 holds row=lane-16, K = {8..15, 24..31}.  `base` -> row start + kk.
__device__ __forceinline__ bf16x16 load_a_frag(const unsigned short* base, int lane) {
  const int k0 = (lane < 16) ? 0 : 8;
  bf16x8 c0 = *(const bf16x8*)(base + k0);
  bf16x8 c1 = *(const bf16x8*)(base + k0 + 16);
  return __builtin_shufflevector(c0, c1,
      0, 1, 2, 3, 4, 5, 6, 7, 8, 9, 10, 11, 12, 13, 14, 15);
}

// B-fragment (32x16 bf16, KxN): lane holds column n=lane&15, K = {0..7, 16..23}
// (lanes 0-15) or {8..15, 24..31} (lanes 16-31).  `colbase` -> column's K-run.
__device__ __forceinline__ bf16x16 load_b_frag(const unsigned short* colbase, int lane) {
  const int k0 = (lane < 16) ? 0 : 8;
  bf16x8 c0 = *(const bf16x8*)(colbase + k0);
  bf16x8 c1 = *(const bf16x8*)(colbase + k0 + 16);
  return __builtin_shufflevector(c0, c1,
      0, 1, 2, 3, 4, 5, 6, 7, 8, 9, 10, 11, 12, 13, 14, 15);
}

__device__ __forceinline__ void wave_lds_fence() {
  asm volatile("s_wait_dscnt 0" ::: "memory");
}

// ---- async global -> LDS copy (async data-mover path, ASYNCcnt) ----
__device__ __forceinline__ void async_b128(const void* g, void* l) {
#if __has_builtin(__builtin_amdgcn_global_load_async_to_lds_b128)
  __builtin_amdgcn_global_load_async_to_lds_b128((v4i*)g, (v4i*)l, 0, 0);
#else
  *(u16x8*)l = *(const u16x8*)g;   // fallback: load + ds_store
#endif
}

__device__ __forceinline__ void wait_async0() {
#if __has_builtin(__builtin_amdgcn_s_wait_asynccnt)
  __builtin_amdgcn_s_wait_asynccnt(0);
#else
  asm volatile("s_wait_asynccnt 0" ::: "memory");
#endif
}

// Stage one contiguous 32 KB A tile (16 rows x 1024 bf16) into LDS.
// 2048 x b128 transfers, 256 threads x 8 each.
__device__ __forceinline__ void stage_a_tile(const unsigned short* gsrc,
                                             unsigned short* lds, int tid) {
#pragma unroll
  for (int i = 0; i < 8; ++i) {
    const size_t off = (size_t)(i * 256 + tid) * 16;   // bytes
    async_b128((const char*)gsrc + off, (char*)lds + off);
  }
  wait_async0();
  __syncthreads();
}

// ---------------------------------------------------------------------------
// Kernel 1: fp32 -> bf16 conversion (grid-stride)
// ---------------------------------------------------------------------------
__global__ void cvt_f32_bf16(const float* __restrict__ src,
                             unsigned short* __restrict__ dst, int n) {
  int i = blockIdx.x * blockDim.x + threadIdx.x;
  int stride = gridDim.x * blockDim.x;
  for (; i < n; i += stride) dst[i] = f32_to_bf16(src[i]);
}

// ---------------------------------------------------------------------------
// Kernel 2: fused QKV projection.  y = x @ W^T + b, W row-major [out,in].
// Block = one (proj, mtile) x 8 n-groups; the 16x1024 A tile is async-staged
// into LDS once and shared by all 8 waves.  One wave -> 16x64 output strip.
// Jobs = 3 * 512 * 16 = 24576; 8 waves/block -> 3072 blocks.
// ---------------------------------------------------------------------------
__global__ __launch_bounds__(256) void qkv_gemm(
    const unsigned short* __restrict__ xb,
    const unsigned short* __restrict__ wb,      // [3][D][D] bf16 (Wq,Wk,Wv)
    const float* __restrict__ bq, const float* __restrict__ bk,
    const float* __restrict__ bv,
    unsigned short* __restrict__ Qh, unsigned short* __restrict__ Kh,
    unsigned short* __restrict__ Vh) {
  __shared__ __align__(16) unsigned short smA[16 * D_MODEL];  // 32 KB

  const int tid  = threadIdx.x;
  const int lane = tid & 31;
  const int wave = tid >> 5;
  const int MT   = NTOK / 16;        // 512
  const int NG   = D_MODEL / 64;     // 16

  const int baseJob = blockIdx.x * 8;          // aligned: same proj/mt in block
  const int proj = baseJob / (MT * NG);
  const int mt   = (baseJob % (MT * NG)) / NG;
  const int ng   = (baseJob % NG) + wave;      // 0..15

  // async-stage the contiguous 16-row activation tile
  stage_a_tile(xb + (size_t)mt * 16 * D_MODEL, smA, tid);

  const unsigned short* w    = wb + (size_t)proj * D_MODEL * D_MODEL;
  const float*          bias = (proj == 0) ? bq : ((proj == 1) ? bk : bv);
  unsigned short*       dst  = (proj == 0) ? Qh : ((proj == 1) ? Kh : Vh);

  const int coln = lane & 15;
  const unsigned short* aptr = smA + (size_t)coln * D_MODEL;  // A row in LDS

  f32x8 acc[4] = {};
  for (int kk = 0; kk < D_MODEL; kk += 32) {
    bf16x16 a = load_a_frag(aptr + kk, lane);   // ds_load_b128 x2
#pragma unroll
    for (int t = 0; t < 4; ++t) {
      const int col = ng * 64 + t * 16 + coln;
      bf16x16 b = load_b_frag(w + (size_t)col * D_MODEL + kk, lane);
      acc[t] = wmma_bf16(a, b, acc[t]);
    }
  }

  const int rowhi = (lane < 16) ? 0 : 8;
#pragma unroll
  for (int t = 0; t < 4; ++t) {
    const int col = ng * 64 + t * 16 + coln;
    const float bcol = bias[col];
    const int h = col >> 6, dh = col & 63;
#pragma unroll
    for (int r = 0; r < 8; ++r) {
      const int tok = mt * 16 + r + rowhi;
      const int bb = tok / SEQ, tt = tok % SEQ;
      const size_t o =
          ((((size_t)bb * N_HEADS + h) * SEQ) + tt) * HEAD_DIM + dh;
      dst[o] = f32_to_bf16(acc[t][r] + bcol);
    }
  }
}

// ---------------------------------------------------------------------------
// Kernel 3: causal flash attention.  One wave per (b, h, 16-query tile).
// Per 32-key step: 4 WMMAs for S = Q K^T, online softmax (lane butterfly),
// P staged through LDS to A-layout, V transposed through LDS to B-layout,
// 4 WMMAs for O += P V.  Y written [tok, D] bf16.
// ---------------------------------------------------------------------------
__global__ __launch_bounds__(256) void attn_fwd(
    const unsigned short* __restrict__ Qh,
    const unsigned short* __restrict__ Kh,
    const unsigned short* __restrict__ Vh,
    unsigned short* __restrict__ Yb) {
  __shared__ __align__(16) unsigned short sm[8 * 2560];  // 40 KB / block
  const int lane = threadIdx.x & 31;
  const int wave = threadIdx.x >> 5;
  unsigned short* Pl = sm + wave * 2560;   // [16][32] bf16 probabilities
  unsigned short* Vt = Pl + 512;           // [64][32] bf16 V transposed

  const int job  = blockIdx.x * 8 + wave;  // 8192 = 4*16*128
  const int qt   = job & 127;
  const int head = job >> 7;
  const int h    = head & 15;
  const int bb   = head >> 4;
  const int q0   = qt * 16;

  const size_t hb = (((size_t)bb * N_HEADS + h) * SEQ) * HEAD_DIM;
  const unsigned short* Qp = Qh + hb;
  const unsigned short* Kp = Kh + hb;
  const unsigned short* Vp = Vh + hb;

  const int coln  = lane & 15;
  const int rowhi = (lane < 16) ? 0 : 8;

  // Q fragments, Dh=64 -> two K=32 chunks, held in registers for all steps
  const int arow = q0 + coln;
  const bf16x16 qa0 = load_a_frag(Qp + (size_t)arow * HEAD_DIM + 0,  lane);
  const bf16x16 qa1 = load_a_frag(Qp + (size_t)arow * HEAD_DIM + 32, lane);

  f32x8 o[4] = {};
  float mrun[8], lrun[8];
#pragma unroll
  for (int r = 0; r < 8; ++r) { mrun[r] = -1e30f; lrun[r] = 0.f; }

  const int kend  = q0 + 16;
  const int npair = (kend + 31) >> 5;

  for (int j = 0; j < npair; ++j) {
    const int kb = j * 32;

    // ---- S = Q K^T for two 16-key tiles (fp32 accum) ----
    f32x8 s0 = {}, s1 = {};
    {
      const unsigned short* kc0 = Kp + (size_t)(kb + coln) * HEAD_DIM;
      const unsigned short* kc1 = Kp + (size_t)(kb + 16 + coln) * HEAD_DIM;
      bf16x16 b00 = load_b_frag(kc0 + 0,  lane);
      bf16x16 b01 = load_b_frag(kc0 + 32, lane);
      bf16x16 b10 = load_b_frag(kc1 + 0,  lane);
      bf16x16 b11 = load_b_frag(kc1 + 32, lane);
      s0 = wmma_bf16(qa0, b00, s0);
      s0 = wmma_bf16(qa1, b01, s0);
      s1 = wmma_bf16(qa0, b10, s1);
      s1 = wmma_bf16(qa1, b11, s1);
    }

    // ---- stage V[32 keys][64 d] transposed into LDS: Vt[d][k] ----
    {
      const unsigned short* vrow = Vp + (size_t)(kb + lane) * HEAD_DIM;
#pragma unroll
      for (int c8 = 0; c8 < 8; ++c8) {
        u16x8 v8 = *(const u16x8*)(vrow + c8 * 8);
#pragma unroll
        for (int e = 0; e < 8; ++e) Vt[(c8 * 8 + e) * 32 + lane] = v8[e];
      }
    }

    // ---- scale + causal mask + online softmax (rows match C layout) ----
    float p0[8], p1[8];
#pragma unroll
    for (int r = 0; r < 8; ++r) {
      const int qrow = q0 + r + rowhi;
      float v0 = s0[r] * 0.125f;               // 1/sqrt(Dh)
      float v1 = s1[r] * 0.125f;
      if (kb + coln      > qrow) v0 = -1e30f;
      if (kb + 16 + coln > qrow) v1 = -1e30f;
      float mt_ = fmaxf(v0, v1);
#pragma unroll
      for (int msk = 1; msk < 16; msk <<= 1)
        mt_ = fmaxf(mt_, __shfl_xor(mt_, msk, 32));
      const float mnew  = fmaxf(mrun[r], mt_);
      const float scale = __expf(mrun[r] - mnew);
      const float e0 = __expf(v0 - mnew);
      const float e1 = __expf(v1 - mnew);
      float ls = e0 + e1;
#pragma unroll
      for (int msk = 1; msk < 16; msk <<= 1) ls += __shfl_xor(ls, msk, 32);
      lrun[r] = lrun[r] * scale + ls;
      mrun[r] = mnew;
      p0[r] = e0;
      p1[r] = e1;
#pragma unroll
      for (int t = 0; t < 4; ++t) o[t][r] *= scale;
    }

    // ---- P to LDS row-major [16][32] bf16 ----
#pragma unroll
    for (int r = 0; r < 8; ++r) {
      const int row = r + rowhi;
      Pl[row * 32 + coln]      = f32_to_bf16(p0[r]);
      Pl[row * 32 + 16 + coln] = f32_to_bf16(p1[r]);
    }
    wave_lds_fence();   // wave-local: all 32 lanes' ds stores visible

    // ---- O += P(16x32) @ V(32x16d) per 16-wide d tile ----
    bf16x16 pa;
    {
      const int k0 = (lane < 16) ? 0 : 8;
      const unsigned short* pp = Pl + coln * 32;   // A row = lane&15
      bf16x8 c0 = *(const bf16x8*)(pp + k0);
      bf16x8 c1 = *(const bf16x8*)(pp + k0 + 16);
      pa = __builtin_shufflevector(c0, c1,
          0, 1, 2, 3, 4, 5, 6, 7, 8, 9, 10, 11, 12, 13, 14, 15);
    }
#pragma unroll
    for (int t = 0; t < 4; ++t) {
      const int dcol = t * 16 + coln;
      bf16x16 vb = load_b_frag(Vt + dcol * 32, lane);
      o[t] = wmma_bf16(pa, vb, o[t]);
    }
  }

  // ---- normalize and store Y[tok][D] ----
#pragma unroll
  for (int r = 0; r < 8; ++r) {
    const float inv = 1.0f / lrun[r];
#pragma unroll
    for (int t = 0; t < 4; ++t) o[t][r] *= inv;
  }
#pragma unroll
  for (int t = 0; t < 4; ++t) {
    const int dcol = h * HEAD_DIM + t * 16 + coln;
#pragma unroll
    for (int r = 0; r < 8; ++r) {
      const int tok = bb * SEQ + q0 + r + rowhi;
      Yb[(size_t)tok * D_MODEL + dcol] = f32_to_bf16(o[t][r]);
    }
  }
}

// ---------------------------------------------------------------------------
// Kernel 4: output projection  out = Y @ Wo^T + bo  (fp32 store)
// Same async-staged-A structure as the QKV GEMM.  1024 blocks.
// ---------------------------------------------------------------------------
__global__ __launch_bounds__(256) void out_gemm(
    const unsigned short* __restrict__ yb,
    const unsigned short* __restrict__ wo,
    const float* __restrict__ bo,
    float* __restrict__ out) {
  __shared__ __align__(16) unsigned short smA[16 * D_MODEL];  // 32 KB

  const int tid  = threadIdx.x;
  const int lane = tid & 31;
  const int wave = tid >> 5;
  const int NG   = D_MODEL / 64;     // 16

  const int baseJob = blockIdx.x * 8;
  const int mt = baseJob / NG;
  const int ng = (baseJob % NG) + wave;

  stage_a_tile(yb + (size_t)mt * 16 * D_MODEL, smA, tid);

  const int coln = lane & 15;
  const unsigned short* aptr = smA + (size_t)coln * D_MODEL;

  f32x8 acc[4] = {};
  for (int kk = 0; kk < D_MODEL; kk += 32) {
    bf16x16 a = load_a_frag(aptr + kk, lane);
#pragma unroll
    for (int t = 0; t < 4; ++t) {
      const int col = ng * 64 + t * 16 + coln;
      bf16x16 b = load_b_frag(wo + (size_t)col * D_MODEL + kk, lane);
      acc[t] = wmma_bf16(a, b, acc[t]);
    }
  }

  const int rowhi = (lane < 16) ? 0 : 8;
#pragma unroll
  for (int t = 0; t < 4; ++t) {
    const int col = ng * 64 + t * 16 + coln;
    const float bcol = bo[col];
#pragma unroll
    for (int r = 0; r < 8; ++r) {
      const int tok = mt * 16 + r + rowhi;
      out[(size_t)tok * D_MODEL + col] = acc[t][r] + bcol;
    }
  }
}

// ---------------------------------------------------------------------------
extern "C" void kernel_launch(void* const* d_in, const int* in_sizes, int n_in,
                              void* d_out, int out_size, void* d_ws,
                              size_t ws_size, hipStream_t stream) {
  const float* x  = (const float*)d_in[0];
  const float* Wq = (const float*)d_in[1];
  const float* bq = (const float*)d_in[2];
  const float* Wk = (const float*)d_in[3];
  const float* bk = (const float*)d_in[4];
  const float* Wv = (const float*)d_in[5];
  const float* bv = (const float*)d_in[6];
  const float* Wo = (const float*)d_in[7];
  const float* bo = (const float*)d_in[8];
  float* out = (float*)d_out;

  const size_t ND  = (size_t)NTOK * D_MODEL;       // 8.39M elems
  const size_t DD  = (size_t)D_MODEL * D_MODEL;    // 1.05M elems

  unsigned short* xb   = (unsigned short*)d_ws;
  unsigned short* wqkv = xb + ND;                  // [3][D][D]
  unsigned short* wob  = wqkv + 3 * DD;
  unsigned short* Qhp  = wob + DD;                 // [B,H,T,Dh]
  unsigned short* Khp  = Qhp + ND;
  unsigned short* Vhp  = Khp + ND;
  unsigned short* Ybp  = Vhp + ND;                 // [tok, D]

  cvt_f32_bf16<<<2048, 256, 0, stream>>>(x, xb, (int)ND);
  cvt_f32_bf16<<<1024, 256, 0, stream>>>(Wq, wqkv + 0 * DD, (int)DD);
  cvt_f32_bf16<<<1024, 256, 0, stream>>>(Wk, wqkv + 1 * DD, (int)DD);
  cvt_f32_bf16<<<1024, 256, 0, stream>>>(Wv, wqkv + 2 * DD, (int)DD);
  cvt_f32_bf16<<<1024, 256, 0, stream>>>(Wo, wob, (int)DD);

  qkv_gemm<<<3072, 256, 0, stream>>>(xb, wqkv, bq, bk, bv, Qhp, Khp, Vhp);
  attn_fwd<<<1024, 256, 0, stream>>>(Qhp, Khp, Vhp, Ybp);
  out_gemm<<<1024, 256, 0, stream>>>(Ybp, wob, bo, out);
}
